// SelfAttention_51505247814326
// MI455X (gfx1250) — matrix operand order repaired and checked by reading, not compiled, not run on hardware
//
#include <hip/hip_runtime.h>

// ---------------------------------------------------------------------------
// Self-attention (B=4, N=4096, D=E=256), fp32 in/out, f16 WMMA internally.
// Double-buffered TDM (tensor_load_to_lds) stages K/V blocks into LDS.
// ---------------------------------------------------------------------------

typedef __attribute__((ext_vector_type(16))) _Float16 v16h;
typedef __attribute__((ext_vector_type(8)))  _Float16 v8h;
typedef __attribute__((ext_vector_type(8)))  float    v8f;
typedef __attribute__((ext_vector_type(4)))  unsigned int u32x4;
typedef __attribute__((ext_vector_type(8)))  int          i32x8;
typedef __attribute__((ext_vector_type(4)))  int          i32x4;

#define BATCH 4
#define SEQ   4096
#define DIM   256
#define ROWS  (BATCH * SEQ)          // 16384

#if defined(__has_builtin)
#if __has_builtin(__builtin_amdgcn_tensor_load_to_lds)
#define HAVE_TDM 1
#endif
#if __has_builtin(__builtin_amdgcn_s_wait_tensorcnt)
#define HAVE_WAIT_TENSOR 1
#endif
#endif
#ifndef HAVE_TDM
#define HAVE_TDM 0
#endif
#ifndef HAVE_WAIT_TENSOR
#define HAVE_WAIT_TENSOR 0
#endif

// ---- WMMA helper ----------------------------------------------------------
__device__ __forceinline__ v8f wmma_f16(v16h a, v16h b, v8f c) {
  return __builtin_amdgcn_wmma_f32_16x16x32_f16(
      /*neg_a=*/false, a, /*neg_b=*/false, b,
      /*c_mod=*/(short)0, c, /*reuse_a=*/false, /*reuse_b=*/false);
}

// A-matrix fragment (16x32 f16) from row-major [16][stride] tile.
// lanes 0-15:  M=lane,     K = {k0..k0+7,  k0+16..k0+23}
// lanes 16-31: M=lane-16,  K = {k0+8..k0+15, k0+24..k0+31}
__device__ __forceinline__ v16h load_a_frag(const _Float16* src, int stride,
                                            int k0, int lane) {
  const int m  = lane & 15;
  const int hi = lane >> 4;
  const _Float16* p = src + (size_t)m * stride + k0 + hi * 8;
  v8h lo = *(const v8h*)(p);
  v8h hh = *(const v8h*)(p + 16);
  v16h r;
#pragma unroll
  for (int i = 0; i < 8; ++i) { r[i] = lo[i]; r[i + 8] = hh[i]; }
  return r;
}

// B-matrix fragment (32x16 f16): src row-major [N][stride], K along the row.
// lanes 0-15 hold K={k0..k0+7, k0+16..k0+23}, lanes 16-31 hold K={k0+8..k0+15, k0+24..k0+31}, N = lane&15.
__device__ __forceinline__ v16h load_b_frag(const _Float16* src, int stride,
                                            int k0, int lane) {
  const int n  = lane & 15;
  const int hi = lane >> 4;
  const _Float16* p = src + (size_t)n * stride + k0 + hi * 8;
  v8h lo = *(const v8h*)(p);
  v8h hh = *(const v8h*)(p + 16);
  v16h r;
#pragma unroll
  for (int i = 0; i < 8; ++i) { r[i] = lo[i]; r[i + 8] = hh[i]; }
  return r;
}

// ---- Tensor Data Mover: 2-D tile load, element size = 2 bytes -------------
#if HAVE_TDM
__device__ __forceinline__ void tdm_load_2d(unsigned lds_byte_off,
                                            const void* gsrc,
                                            unsigned tile_w, unsigned tile_h,
                                            unsigned tensor_w, unsigned tensor_h,
                                            unsigned row_stride_elems) {
  unsigned long long ga = (unsigned long long)gsrc;
  // D# group0: count=1 | lds_addr | global_addr[56:0] | type=2
  u32x4 g0 = {1u, lds_byte_off, (unsigned)(ga & 0xFFFFFFFFu),
              (unsigned)(((ga >> 32) & 0x01FFFFFFu) | (2u << 30))};
  // D# group1 (bit-packed per CDNA5 ISA), data_size=1 (2B)
  i32x8 g1 = {(int)0x00010000,
              (int)((tensor_w & 0xFFFFu) << 16),                            // tensor_dim0[15:0]
              (int)(((tensor_w >> 16) & 0xFFFFu) | ((tensor_h & 0xFFFFu) << 16)),
              (int)(((tensor_h >> 16) & 0xFFFFu) | ((tile_w & 0xFFFFu) << 16)),
              (int)(tile_h & 0xFFFFu),                                      // tile_dim1, tile_dim2=0
              (int)row_stride_elems,                                        // tensor_dim0_stride lo
              0, 0};
  i32x4 z4 = {0, 0, 0, 0};
#if __clang_major__ >= 23
  i32x8 z8 = {0, 0, 0, 0, 0, 0, 0, 0};
  __builtin_amdgcn_tensor_load_to_lds(g0, g1, z4, z4, z8, 0);
#else
  __builtin_amdgcn_tensor_load_to_lds(g0, g1, z4, z4, 0);
#endif
}

__device__ __forceinline__ void wait_tensorcnt0() {
#if HAVE_WAIT_TENSOR
  __builtin_amdgcn_s_wait_tensorcnt(0);
#else
  asm volatile("s_wait_tensorcnt 0x0" ::: "memory");
#endif
}
#endif  // HAVE_TDM

// ---------------------------------------------------------------------------
// Kernel 0: fp32 -> f16 conversion; W transposed so contraction dim is
// contiguous per output column (B-operand friendly).
// ---------------------------------------------------------------------------
__global__ void __launch_bounds__(256) convert_kernel(
    const float* __restrict__ x,  const float* __restrict__ Wq,
    const float* __restrict__ Wk, const float* __restrict__ Wv,
    _Float16* __restrict__ Xh, _Float16* __restrict__ Wth) {
  const int tid = blockIdx.x * blockDim.x + threadIdx.x;
  const int nth = gridDim.x * blockDim.x;
  for (int i = tid; i < ROWS * DIM; i += nth) Xh[i] = (_Float16)x[i];
  for (int i = tid; i < 3 * DIM * DIM; i += nth) {
    const int mat = i >> 16;
    const int n   = (i >> 8) & 255;
    const int k   = i & 255;
    const float* W = (mat == 0) ? Wq : (mat == 1) ? Wk : Wv;
    Wth[i] = (_Float16)W[k * DIM + n];   // Wth[mat][n][k]
  }
}

// ---------------------------------------------------------------------------
// Kernel 1: QKV projection. One block per 16-row tile; 8 waves x 6 tiles.
// B fragments flow through a depth-2 rotating register pipeline so one
// load pair stays in flight behind each WMMA instead of load->wait0->wmma.
// ---------------------------------------------------------------------------
__global__ void __launch_bounds__(256) qkv_kernel(
    const _Float16* __restrict__ Xh, const _Float16* __restrict__ Wth,
    const float* __restrict__ bq, const float* __restrict__ bk,
    const float* __restrict__ bv,
    _Float16* __restrict__ Qh, _Float16* __restrict__ Kh,
    _Float16* __restrict__ Vth) {
  const int lane  = threadIdx.x & 31;
  const int wave  = threadIdx.x >> 5;
  const int hi    = lane >> 4;
  const int row0  = blockIdx.x * 16;
  const int batch = row0 >> 12;
  const int mrow  = row0 & (SEQ - 1);
  const _Float16* xa = Xh + (size_t)row0 * DIM;

  v16h af[8];
#pragma unroll
  for (int dk = 0; dk < 8; ++dk) af[dk] = load_a_frag(xa, DIM, dk * 32, lane);

#pragma unroll 1
  for (int t = wave * 6; t < wave * 6 + 6; ++t) {   // t in [0,48)
    const int mat = t >> 4;       // 0:Q 1:K 2:V
    const int col = t & 15;
    const _Float16* wb = Wth + mat * (DIM * DIM) + (size_t)(col * 16) * DIM;

    v16h bb[2];
    bb[0] = load_b_frag(wb, DIM, 0, lane);
    v8f c = {};
#pragma unroll
    for (int dk = 0; dk < 8; ++dk) {
      if (dk < 7) bb[(dk + 1) & 1] = load_b_frag(wb, DIM, (dk + 1) * 32, lane);
      c = wmma_f16(af[dk], bb[dk & 1], c);
    }

    const float* bias = (mat == 0) ? bq : (mat == 1) ? bk : bv;
    const float bval = bias[col * 16 + (lane & 15)];
    if (mat < 2) {
      _Float16* dst = (mat == 0) ? Qh : Kh;
#pragma unroll
      for (int r = 0; r < 8; ++r)
        dst[(size_t)(row0 + r + hi * 8) * DIM + col * 16 + (lane & 15)] =
            (_Float16)(c[r] + bval);
    } else {
#pragma unroll
      for (int r = 0; r < 8; ++r)
        Vth[((size_t)batch * DIM + col * 16 + (lane & 15)) * SEQ +
            (mrow + r + hi * 8)] = (_Float16)(c[r] + bval);
    }
  }
}

// ---------------------------------------------------------------------------
// Kernel 2: flash attention. Block = 4 waves = 4 q-tiles (64 q rows), same
// batch. K (32x256) and V (256x32) blocks stream through DOUBLE-BUFFERED LDS
// via the Tensor Data Mover: per iteration, wait on the current buffer's
// pair, one barrier, issue the next block's pair into the other buffer, then
// 16 WMMA (S=QK^T) + online softmax + 16 WMMA (O+=PV) out of LDS -- the DMA
// for block kb+1 overlaps the compute of block kb.
// ---------------------------------------------------------------------------
#define AW 4                                  // waves per block
#define BUF_HALFS (2 * 32 * DIM)              // K+V per buffer, in halves (32KB)
#define LDS_P_OFF (2 * BUF_HALFS)             // element offset of P region
#define SMEM_BYTES ((2 * BUF_HALFS + AW * 16 * 32) * 2)   // 68 KB

__global__ void __launch_bounds__(32 * AW) attn_kernel(
    const _Float16* __restrict__ Qh, const _Float16* __restrict__ Kh,
    const _Float16* __restrict__ Vth, float* __restrict__ out) {
  extern __shared__ _Float16 smem[];
  _Float16* ldsP = smem + LDS_P_OFF;          // [AW][16*32]

  const int lane  = threadIdx.x & 31;
  const int wave  = threadIdx.x >> 5;
  const int hi    = lane >> 4;
  const int g     = blockIdx.x * AW + wave;   // q-tile id [0,1024)
  const int batch = g >> 8;
  const int q0    = (g & 255) * 16;

  const _Float16* Qg = Qh + ((size_t)batch * SEQ + q0) * DIM;
  const _Float16* Kb = Kh + (size_t)batch * SEQ * DIM;
  const _Float16* Vb = Vth + (size_t)batch * DIM * SEQ;

  // Q tile lives in registers: reused for all 128 key blocks.
  v16h qf[8];
#pragma unroll
  for (int dk = 0; dk < 8; ++dk) qf[dk] = load_a_frag(Qg, DIM, dk * 32, lane);

  v8f acc[16];
#pragma unroll
  for (int i = 0; i < 16; ++i) acc[i] = (v8f){0, 0, 0, 0, 0, 0, 0, 0};
  float mrow[8], lrow[8];
#pragma unroll
  for (int r = 0; r < 8; ++r) { mrow[r] = -1e30f; lrow[r] = 0.f; }
  const float sc = 0.0625f;   // 1/sqrt(256)

  // helper lambda: issue TDM (or fallback copy) of key-block kb into buf
  auto stage = [&](int kb, int buf) {
#if HAVE_TDM
    if (wave == 0) {
      const unsigned base = (unsigned)buf * (BUF_HALFS * 2);   // bytes
      // K block: 32 rows x 256 cols, row stride 256 elems
      tdm_load_2d(base, Kb + (size_t)kb * 32 * DIM, DIM, 32, DIM, SEQ, DIM);
      // V block: 256 rows x 32 cols out of Vt[256][4096], row stride 4096
      tdm_load_2d(base + 32 * DIM * 2, Vb + kb * 32, 32, DIM, SEQ, DIM, SEQ);
    }
#else
    _Float16* dK = smem + buf * BUF_HALFS;
    _Float16* dV = dK + 32 * DIM;
    for (int i = threadIdx.x; i < 1024; i += 32 * AW)
      *(v8h*)(dK + i * 8) = *(const v8h*)(Kb + (size_t)kb * 32 * DIM + i * 8);
    for (int i = threadIdx.x; i < 1024; i += 32 * AW) {
      const int d = i >> 2, c8 = (i & 3) * 8;
      *(v8h*)(dV + d * 32 + c8) = *(const v8h*)(Vb + (size_t)d * SEQ + kb * 32 + c8);
    }
#endif
  };

  stage(0, 0);   // prologue: fill buffer 0

  for (int kb = 0; kb < SEQ / 32; ++kb) {
    const int cur = kb & 1;
#if HAVE_TDM
    if (wave == 0) wait_tensorcnt0();        // current buffer's pair complete
#endif
    __syncthreads();                         // data visible; prev reads done
    if (kb + 1 < SEQ / 32) stage(kb + 1, cur ^ 1);   // DMA overlaps compute

    const _Float16* ldsK = smem + cur * BUF_HALFS;        // [32][256]
    const _Float16* ldsV = ldsK + 32 * DIM;               // [256][32]

    // ---- S = Q K^T for 32 keys (two 16x16 tiles), B operands from LDS -----
    v8f s0 = {}, s1 = {};
    v16h k0b[2], k1b[2];
    k0b[0] = load_b_frag(ldsK, DIM, 0, lane);
    k1b[0] = load_b_frag(ldsK + 16 * DIM, DIM, 0, lane);
#pragma unroll
    for (int dk = 0; dk < 8; ++dk) {
      if (dk < 7) {
        k0b[(dk + 1) & 1] = load_b_frag(ldsK, DIM, (dk + 1) * 32, lane);
        k1b[(dk + 1) & 1] = load_b_frag(ldsK + 16 * DIM, DIM, (dk + 1) * 32, lane);
      }
      s0 = wmma_f16(qf[dk], k0b[dk & 1], s0);
      s1 = wmma_f16(qf[dk], k1b[dk & 1], s1);
    }

    // ---- online softmax (row = r + 8*hi, spread over 16 lanes) ------------
    float corr[8];
#pragma unroll
    for (int r = 0; r < 8; ++r) {
      const float a0 = s0[r] * sc, a1 = s1[r] * sc;
      float mb = fmaxf(a0, a1);
#pragma unroll
      for (int off = 1; off < 16; off <<= 1)
        mb = fmaxf(mb, __shfl_xor(mb, off, 32));
      const float mn = fmaxf(mrow[r], mb);
      corr[r] = __expf(mrow[r] - mn);
      const float p0 = __expf(a0 - mn);
      const float p1 = __expf(a1 - mn);
      s0[r] = p0; s1[r] = p1;
      float ls = p0 + p1;
#pragma unroll
      for (int off = 1; off < 16; off <<= 1)
        ls += __shfl_xor(ls, off, 32);
      lrow[r] = lrow[r] * corr[r] + ls;
      mrow[r] = mn;
    }
#pragma unroll
    for (int ct = 0; ct < 16; ++ct)
#pragma unroll
      for (int r = 0; r < 8; ++r) acc[ct][r] *= corr[r];

    // ---- D-layout P -> wave-private LDS (16x32) -> A-fragment -------------
    _Float16* pw = ldsP + wave * (16 * 32);
#pragma unroll
    for (int r = 0; r < 8; ++r) {
      const int row = r + hi * 8;
      pw[row * 32 + (lane & 15)]      = (_Float16)s0[r];
      pw[row * 32 + 16 + (lane & 15)] = (_Float16)s1[r];
    }
    v16h ap = load_a_frag(pw, 32, 0, lane);

    // ---- O += P V (V block resident in LDS, m contiguous) -----------------
    v16h vb2[2];
    vb2[0] = load_b_frag(ldsV, 32, 0, lane);
#pragma unroll
    for (int ct = 0; ct < 16; ++ct) {
      if (ct < 15)
        vb2[(ct + 1) & 1] = load_b_frag(ldsV + (ct + 1) * 16 * 32, 32, 0, lane);
      acc[ct] = wmma_f16(ap, vb2[ct & 1], acc[ct]);
    }
  }

  // ---- epilogue: normalize and store fp32 ---------------------------------
  float* outg = out + ((size_t)batch * SEQ + q0) * DIM;
#pragma unroll
  for (int r = 0; r < 8; ++r) {
    const float inv = 1.f / lrow[r];
#pragma unroll
    for (int ct = 0; ct < 16; ++ct)
      outg[(size_t)(r + hi * 8) * DIM + ct * 16 + (lane & 15)] = acc[ct][r] * inv;
  }
}

// ---------------------------------------------------------------------------
extern "C" void kernel_launch(void* const* d_in, const int* in_sizes, int n_in,
                              void* d_out, int out_size, void* d_ws,
                              size_t ws_size, hipStream_t stream) {
  const float* x  = (const float*)d_in[0];
  const float* Wq = (const float*)d_in[1];
  const float* bq = (const float*)d_in[2];
  const float* Wk = (const float*)d_in[3];
  const float* bk = (const float*)d_in[4];
  const float* Wv = (const float*)d_in[5];
  const float* bv = (const float*)d_in[6];
  float* out = (float*)d_out;

  char* ws = (char*)d_ws;
  size_t off = 0;
  auto take = [&](size_t bytes) {
    char* p = ws + off;
    off = (off + bytes + 255) & ~(size_t)255;
    return p;
  };
  _Float16* Xh  = (_Float16*)take((size_t)ROWS * DIM * 2);       // 8 MB
  _Float16* Wth = (_Float16*)take((size_t)3 * DIM * DIM * 2);    // 384 KB
  _Float16* Qh  = (_Float16*)take((size_t)ROWS * DIM * 2);       // 8 MB
  _Float16* Kh  = (_Float16*)take((size_t)ROWS * DIM * 2);       // 8 MB
  _Float16* Vth = (_Float16*)take((size_t)ROWS * DIM * 2);       // 8 MB
  (void)ws_size; (void)in_sizes; (void)n_in; (void)out_size;

  convert_kernel<<<2048, 256, 0, stream>>>(x, Wq, Wk, Wv, Xh, Wth);
  qkv_kernel<<<ROWS / 16, 256, 0, stream>>>(Xh, Wth, bq, bk, bv, Qh, Kh, Vth);
  attn_kernel<<<(ROWS / 16) / AW, 32 * AW, SMEM_BYTES, stream>>>(Qh, Kh, Vth, out);
}
